// GNN_36077725286460
// MI455X (gfx1250) — matrix operand order repaired and checked by reading, not compiled, not run on hardware
//
#include <hip/hip_runtime.h>

// ---------------------------------------------------------------------------
// Problem constants (match reference)
// ---------------------------------------------------------------------------
#define BB   8
#define CC   192
#define HH   56
#define WW   56
#define NN   (HH * WW)      // 3136
#define KK   9
#define C2   (2 * CC)       // 384
#define OUTC 192
#define MM   (BB * NN)      // 25088 rows
#define EPSF 1e-5f

typedef __bf16 bf16x16 __attribute__((ext_vector_type(16)));
typedef __bf16 bf16x8  __attribute__((ext_vector_type(8)));
typedef float  f32x8   __attribute__((ext_vector_type(8)));

__device__ __forceinline__ float gelu_exact(float v) {
    return 0.5f * v * (1.0f + erff(v * 0.70710678118654752f));
}

// ---------------------------------------------------------------------------
// 1) x [B,C,N] f32  ->  xt [B,N,C] bf16   (tiled LDS transpose, both coalesced)
//    grid(98, 6, 8) block(32, 8)
// ---------------------------------------------------------------------------
__global__ void k_xt(const float* __restrict__ x, __bf16* __restrict__ xtb) {
    __shared__ float t[32][33];
    const int b  = blockIdx.z;
    const int n0 = blockIdx.x * 32;
    const int c0 = blockIdx.y * 32;
    const int tx = threadIdx.x, ty = threadIdx.y;
    #pragma unroll
    for (int i = 0; i < 32; i += 8)
        t[ty + i][tx] = x[((size_t)(b * CC + c0 + ty + i)) * NN + n0 + tx];
    __syncthreads();
    #pragma unroll
    for (int i = 0; i < 32; i += 8)
        xtb[((size_t)b * NN + n0 + ty + i) * CC + c0 + tx] = (__bf16)t[tx][ty + i];
}

// ---------------------------------------------------------------------------
// 2) gather + max-over-K + channel interleave -> feats [M, 2C] bf16
//    grid(M) block(192);  gathers are 384B-contiguous per neighbor (L2 hits)
// ---------------------------------------------------------------------------
__global__ void k_feats(const __bf16* __restrict__ xtb,
                        const int* __restrict__ eidx,
                        __bf16* __restrict__ feats) {
    const int row = blockIdx.x;          // b*N + n
    const int b = row / NN, n = row % NN;
    const int c = threadIdx.x;           // 0..191
    const int* e0 = eidx + ((size_t)(0 * BB + b) * NN + n) * KK;
    const int* e1 = eidx + ((size_t)(1 * BB + b) * NN + n) * KK;
    const __bf16* base = xtb + (size_t)b * NN * CC;
    float m = -__builtin_inff();
    #pragma unroll
    for (int k = 0; k < KK; ++k) {
        const int j = e0[k], i = e1[k];
        const float d = (float)base[(size_t)j * CC + c] - (float)base[(size_t)i * CC + c];
        m = fmaxf(m, d);
    }
    feats[(size_t)row * C2 + 2 * c]     = base[(size_t)n * CC + c];
    feats[(size_t)row * C2 + 2 * c + 1] = (__bf16)m;
}

// ---------------------------------------------------------------------------
// 3) weight repack: wt[c][o] = bf16(w[o][c])  (tiny; makes B-operand loads
//    contiguous per k-row)
// ---------------------------------------------------------------------------
__global__ void k_wt(const float* __restrict__ w, __bf16* __restrict__ wt,
                     int O, int Cin) {
    int idx = blockIdx.x * blockDim.x + threadIdx.x;
    if (idx >= O * Cin) return;
    const int o = idx % O, c = idx / O;
    wt[(size_t)c * O + o] = (__bf16)w[(size_t)o * Cin + c];
}

// ---------------------------------------------------------------------------
// 4) WMMA GEMM:  Cout[M,Ncols] = A[M,384](bf16) * Bt[384,Ncols](bf16) + bias
//    block = 128 (4 waves); block computes 64 rows x 64 cols.
//    B slice [384 x 64] is staged once into LDS (144B row pitch: 16B-aligned,
//    no systematic bank conflict for lane distance < 32), shared by all waves.
//    A operand is register double-buffered: k+1 loads issue before tile-k WMMAs.
//    grid(M/64, Ncols/64)
// ---------------------------------------------------------------------------
#define BPITCH 72   // bf16 elements per LDS B-row (144 bytes)

__global__ void __launch_bounds__(128)
k_gemm(const __bf16* __restrict__ A, const __bf16* __restrict__ Bw,
       const float* __restrict__ bias, float* __restrict__ Cout,
       int Ncols) {
    constexpr int KD = C2;                       // 384
    __shared__ __bf16 ldsB[KD * BPITCH];         // 54 KB

    const int tid  = threadIdx.x;
    const int lane = tid & 31;
    const int wave = tid >> 5;
    const int m0   = (blockIdx.x * 4 + wave) * 16;
    const int n0   = blockIdx.y * 64;

    // ---- stage B slice into LDS: 384 rows x 64 bf16 (8 x 16B chunks/row) ----
    for (int i = tid; i < KD * 8; i += 128) {
        const int r = i >> 3, q = i & 7;
        const float4 v = *(const float4*)(Bw + (size_t)r * Ncols + n0 + q * 8);
        *(float4*)(ldsB + r * BPITCH + q * 8) = v;
    }
    __syncthreads();

    // ---- main loop: A double-buffered in registers, B from LDS ----
    const int half = lane >> 4;
    const __bf16* arow = A + (size_t)(m0 + (lane & 15)) * KD + 8 * half;

    union AV { bf16x16 v; bf16x8 h[2]; };
    AV acur, anext;
    acur.h[0] = *(const bf16x8*)(arow);
    acur.h[1] = *(const bf16x8*)(arow + 16);

    f32x8 acc0 = {}, acc1 = {}, acc2 = {}, acc3 = {};

    for (int kt = 0; kt < KD / 32; ++kt) {
        if (kt < KD / 32 - 1) {                  // prefetch next A tile
            anext.h[0] = *(const bf16x8*)(arow + 32 * (kt + 1));
            anext.h[1] = *(const bf16x8*)(arow + 32 * (kt + 1) + 16);
        }
        const __bf16* bl = ldsB + (size_t)(kt * 32 + 8 * half) * BPITCH + (lane & 15);
        bf16x16 b0, b1, b2, b3;
        #pragma unroll
        for (int s = 0; s < 16; ++s) {
            const int kk = (s < 8) ? s : s + 8;
            b0[s] = bl[kk * BPITCH];
            b1[s] = bl[kk * BPITCH + 16];
            b2[s] = bl[kk * BPITCH + 32];
            b3[s] = bl[kk * BPITCH + 48];
        }
        acc0 = __builtin_amdgcn_wmma_f32_16x16x32_bf16(false, acur.v, false, b0, (short)0, acc0, false, false);
        acc1 = __builtin_amdgcn_wmma_f32_16x16x32_bf16(false, acur.v, false, b1, (short)0, acc1, false, false);
        acc2 = __builtin_amdgcn_wmma_f32_16x16x32_bf16(false, acur.v, false, b2, (short)0, acc2, false, false);
        acc3 = __builtin_amdgcn_wmma_f32_16x16x32_bf16(false, acur.v, false, b3, (short)0, acc3, false, false);
        acur = anext;
    }

    // ---- epilogue: D layout: VGPR r -> row (r + 8*half), lane%16 -> column ----
    const int ncol = lane & 15;
    f32x8 accs[4] = {acc0, acc1, acc2, acc3};
    #pragma unroll
    for (int j = 0; j < 4; ++j) {
        const int col = n0 + 16 * j + ncol;
        const float bz = bias[col];
        #pragma unroll
        for (int r = 0; r < 8; ++r)
            Cout[(size_t)(m0 + r + 8 * half) * Ncols + col] = accs[j][r] + bz;
    }
}

// ---------------------------------------------------------------------------
// 5) BN batch-stats, stage 1: per-(row-chunk, channel) partial sums.
//    block = Ncols threads (channel = tid), fully coalesced. grid = M/64.
// ---------------------------------------------------------------------------
__global__ void k_stats_part(const float* __restrict__ h,
                             float* __restrict__ psum, float* __restrict__ psq,
                             int Ncols) {
    const int c  = threadIdx.x;
    const int r0 = blockIdx.x * 64;
    float s = 0.f, q = 0.f;
    for (int r = 0; r < 64; ++r) {
        const float v = h[(size_t)(r0 + r) * Ncols + c];
        s += v; q += v * v;
    }
    psum[(size_t)blockIdx.x * Ncols + c] = s;
    psq [(size_t)blockIdx.x * Ncols + c] = q;
}

// stage 2: fixed-order reduce -> scale/shift  (deterministic, no atomics)
__global__ void k_stats_final(const float* __restrict__ psum, const float* __restrict__ psq,
                              const float* __restrict__ gamma, const float* __restrict__ beta,
                              float* __restrict__ scale, float* __restrict__ shift,
                              int Ncols, int nblk) {
    const int c = threadIdx.x;
    float s = 0.f, q = 0.f;
    for (int i = 0; i < nblk; ++i) {
        s += psum[(size_t)i * Ncols + c];
        q += psq [(size_t)i * Ncols + c];
    }
    const float inv  = 1.0f / (float)MM;
    const float mean = s * inv;
    const float var  = q * inv - mean * mean;
    const float sc   = gamma[c] * rsqrtf(var + EPSF);
    scale[c] = sc;
    shift[c] = beta[c] - mean * sc;
}

// ---------------------------------------------------------------------------
// 6) BN-apply + exact GELU -> bf16 activations (input to GEMM2)
// ---------------------------------------------------------------------------
__global__ void k_apply_bf16(const float* __restrict__ h,
                             const float* __restrict__ scale, const float* __restrict__ shift,
                             __bf16* __restrict__ outb, int Ncols, long long total) {
    const long long idx = (long long)blockIdx.x * blockDim.x + threadIdx.x;
    if (idx >= total) return;
    const int c = (int)(idx % Ncols);
    const float v = h[idx] * scale[c] + shift[c];
    outb[idx] = (__bf16)gelu_exact(v);
}

// ---------------------------------------------------------------------------
// 7) BN-apply + GELU + NCHW transpose -> d_out [B, OUT, N] f32
//    tiled LDS transpose: reads coalesced over channels, writes coalesced over n
//    grid(98, 6, 8) block(32, 8)
// ---------------------------------------------------------------------------
__global__ void k_out(const float* __restrict__ h2,
                      const float* __restrict__ scale, const float* __restrict__ shift,
                      float* __restrict__ out) {
    __shared__ float t[32][33];
    const int b  = blockIdx.z;
    const int n0 = blockIdx.x * 32;
    const int o0 = blockIdx.y * 32;
    const int tx = threadIdx.x, ty = threadIdx.y;
    #pragma unroll
    for (int i = 0; i < 32; i += 8) {
        const int o = o0 + tx;
        const float v = h2[((size_t)(b * NN + n0 + ty + i)) * OUTC + o] * scale[o] + shift[o];
        t[ty + i][tx] = gelu_exact(v);
    }
    __syncthreads();
    #pragma unroll
    for (int i = 0; i < 32; i += 8)
        out[((size_t)(b * OUTC + o0 + ty + i)) * NN + n0 + tx] = t[tx][ty + i];
}

// ---------------------------------------------------------------------------
// Host side
// ---------------------------------------------------------------------------
extern "C" void kernel_launch(void* const* d_in, const int* in_sizes, int n_in,
                              void* d_out, int out_size, void* d_ws, size_t ws_size,
                              hipStream_t stream) {
    const float* x     = (const float*)d_in[0];
    const int*   eidx  = (const int*)  d_in[1];
    const float* w1    = (const float*)d_in[2];
    const float* b1    = (const float*)d_in[3];
    const float* g1    = (const float*)d_in[4];
    const float* beta1 = (const float*)d_in[5];
    const float* w2    = (const float*)d_in[6];
    const float* b2    = (const float*)d_in[7];
    const float* g2    = (const float*)d_in[8];
    const float* beta2 = (const float*)d_in[9];
    float* out = (float*)d_out;

    // workspace carve-up (all sizes are multiples of 256B)
    char* ws = (char*)d_ws;
    size_t off = 0;
    auto take = [&](size_t bytes) { char* p = ws + off; off += (bytes + 255) & ~(size_t)255; return p; };
    __bf16* xtb    = (__bf16*)take((size_t)MM * CC * 2);      //  9.6 MB
    __bf16* feats  = (__bf16*)take((size_t)MM * C2 * 2);      // 19.3 MB
    __bf16* w1t    = (__bf16*)take((size_t)C2 * C2 * 2);
    __bf16* w2t    = (__bf16*)take((size_t)C2 * OUTC * 2);
    float*  h1     = (float*) take((size_t)MM * C2 * 4);      // 38.5 MB
    __bf16* act1   = (__bf16*)take((size_t)MM * C2 * 2);      // 19.3 MB
    float*  h2     = (float*) take((size_t)MM * OUTC * 4);    // 19.3 MB
    const int nblk = MM / 64;                                 // 392
    float* psum1   = (float*)take((size_t)nblk * C2 * 4);
    float* psq1    = (float*)take((size_t)nblk * C2 * 4);
    float* psum2   = (float*)take((size_t)nblk * OUTC * 4);
    float* psq2    = (float*)take((size_t)nblk * OUTC * 4);
    float* scale1  = (float*)take(C2 * 4);
    float* shift1  = (float*)take(C2 * 4);
    float* scale2  = (float*)take(OUTC * 4);
    float* shift2  = (float*)take(OUTC * 4);
    (void)ws_size; (void)in_sizes; (void)n_in; (void)out_size;

    // 1) transpose x -> xt (bf16)
    k_xt<<<dim3(NN / 32, CC / 32, BB), dim3(32, 8), 0, stream>>>(x, xtb);
    // 2) gather + max + interleave -> feats
    k_feats<<<dim3(MM), dim3(CC), 0, stream>>>(xtb, eidx, feats);
    // 3) weight repacks
    k_wt<<<dim3((C2 * C2 + 255) / 256), dim3(256), 0, stream>>>(w1, w1t, C2, C2);
    k_wt<<<dim3((OUTC * C2 + 255) / 256), dim3(256), 0, stream>>>(w2, w2t, OUTC, C2);
    // 4) GEMM1: h1 = feats * w1^T + b1
    k_gemm<<<dim3(MM / 64, C2 / 64), dim3(128), 0, stream>>>(feats, w1t, b1, h1, C2);
    // 5) BN1 stats
    k_stats_part <<<dim3(nblk), dim3(C2), 0, stream>>>(h1, psum1, psq1, C2);
    k_stats_final<<<dim3(1), dim3(C2), 0, stream>>>(psum1, psq1, g1, beta1, scale1, shift1, C2, nblk);
    // 6) BN1-apply + GELU -> bf16
    {
        const long long total = (long long)MM * C2;
        k_apply_bf16<<<dim3((unsigned)((total + 255) / 256)), dim3(256), 0, stream>>>(
            h1, scale1, shift1, act1, C2, total);
    }
    // 7) GEMM2: h2 = act1 * w2^T + b2
    k_gemm<<<dim3(MM / 64, OUTC / 64), dim3(128), 0, stream>>>(act1, w2t, b2, h2, OUTC);
    // 8) BN2 stats
    k_stats_part <<<dim3(nblk), dim3(OUTC), 0, stream>>>(h2, psum2, psq2, OUTC);
    k_stats_final<<<dim3(1), dim3(OUTC), 0, stream>>>(psum2, psq2, g2, beta2, scale2, shift2, OUTC, nblk);
    // 9) BN2-apply + GELU + NCHW transpose -> out
    k_out<<<dim3(NN / 32, OUTC / 32, BB), dim3(32, 8), 0, stream>>>(h2, scale2, shift2, out);
}
